// MoEBlock_76596446757300
// MI455X (gfx1250) — matrix operand order repaired and checked by reading, not compiled, not run on hardware
//
#include <hip/hip_runtime.h>
#include <hip/hip_bf16.h>
#include <math.h>

// Problem dims (from reference): B=2, T=2048, C=1024, H=4096, E=8, RED=16
constexpr int kS   = 4096;   // B*T tokens
constexpr int kC   = 1024;
constexpr int kH   = 4096;
constexpr int kE   = 8;
constexpr int kRED = 16;

constexpr int kPitch = 40;   // bf16 pitch for LDS fragment rows (16B aligned, low-conflict)

typedef __attribute__((ext_vector_type(16))) __bf16 v16bf;
typedef __attribute__((ext_vector_type(8)))  __bf16 v8bf;
typedef __attribute__((ext_vector_type(8)))  float  v8f;

__device__ __forceinline__ unsigned short f2bf_bits(float f) {
    union { __bf16 b; unsigned short s; } cv; cv.b = (__bf16)f; return cv.s;
}
__device__ __forceinline__ unsigned pack2bf(float lo, float hi) {
    return (unsigned)f2bf_bits(lo) | ((unsigned)f2bf_bits(hi) << 16);
}

// ---- CDNA5 async global->LDS copy (16B per lane), tracked on ASYNCcnt ------
__device__ __forceinline__ unsigned lds_byte_off(const void* p) {
#if defined(__HIP_DEVICE_COMPILE__)
    return (unsigned)(size_t)(const __attribute__((address_space(3))) void*)p;
#else
    return 0;
#endif
}
__device__ __forceinline__ void async_copy_b128(unsigned lds_off, const void* gptr) {
#if defined(__HIP_DEVICE_COMPILE__)
    asm volatile("global_load_async_to_lds_b128 %0, %1, off"
                 :: "v"(lds_off), "v"((unsigned long long)(size_t)gptr)
                 : "memory");
#else
    (void)lds_off; (void)gptr;
#endif
}
__device__ __forceinline__ void async_wait0() {
#if defined(__HIP_DEVICE_COMPILE__)
    asm volatile("s_wait_asynccnt 0x0" ::: "memory");
#endif
}

// ---------------------------------------------------------------------------
// 1) Gating: red = x @ wr^T -> logits = red @ normalize(wg)^T -> top-1.
// ---------------------------------------------------------------------------
__global__ __launch_bounds__(256) void gate_kernel(
    const float* __restrict__ x, const float* __restrict__ wr,
    const float* __restrict__ wg, float* __restrict__ scores,
    int* __restrict__ eidx) {
    __shared__ float lwg[kE * kRED];
    int tid = threadIdx.x;
    if (tid < kE * kRED) lwg[tid] = wg[tid];
    __syncthreads();

    int wave = tid >> 5, lane = tid & 31;
    int s = blockIdx.x * 8 + wave;

    float acc[kRED];
#pragma unroll
    for (int r = 0; r < kRED; ++r) acc[r] = 0.f;

    const float* xr = x + (size_t)s * kC;
    for (int i = 0; i < kC / 32; ++i) {
        int k = lane + i * 32;
        float xv = xr[k];
#pragma unroll
        for (int r = 0; r < kRED; ++r) acc[r] += xv * wr[r * kC + k];
    }
#pragma unroll
    for (int r = 0; r < kRED; ++r) {
#pragma unroll
        for (int m = 16; m >= 1; m >>= 1) acc[r] += __shfl_xor(acc[r], m, 32);
    }
    if (lane == 0) {
        float mx = -1e30f; int best = 0; float lg[kE];
#pragma unroll
        for (int e = 0; e < kE; ++e) {
            float nrm = 0.f, dot = 0.f;
#pragma unroll
            for (int r = 0; r < kRED; ++r) {
                float w = lwg[e * kRED + r];
                nrm += w * w; dot += acc[r] * w;
            }
            nrm = fmaxf(sqrtf(nrm), 1e-4f);
            float l = dot / nrm;
            lg[e] = l;
            if (l > mx) { mx = l; best = e; }   // first-max tiebreak like argmax
        }
        float den = 0.f;
#pragma unroll
        for (int e = 0; e < kE; ++e) den += expf(lg[e] - mx);
        scores[s] = 1.f / den;   // max of softmax
        eidx[s]   = best;
    }
}

// ---------------------------------------------------------------------------
// 2) Deterministic stable scatter: per-expert token lists + counts.
// ---------------------------------------------------------------------------
__global__ void scatter_kernel(const int* __restrict__ eidx,
                               int* __restrict__ counts,
                               int* __restrict__ list) {
    if (threadIdx.x == 0) {
        int cnt[kE];
        for (int e = 0; e < kE; ++e) cnt[e] = 0;
        for (int s = 0; s < kS; ++s) {
            int e = eidx[s];
            list[e * kS + cnt[e]] = s;
            cnt[e]++;
        }
        for (int e = 0; e < kE; ++e) counts[e] = cnt[e];
    }
}

// ---------------------------------------------------------------------------
// 3) fc1: h = gelu(x @ w1[e] + b1[e]) -> bf16, grouped by expert.
//    Block tile 16 tokens x 256 H-cols, 8 waves; each wave: 2 WMMA / K-step.
//    LDS holds bf16 in fragment order: A row-pitch 40, B transposed per-n.
// ---------------------------------------------------------------------------
__global__ __launch_bounds__(256) void fc1_kernel(
    const float* __restrict__ x, const float* __restrict__ w1,
    const float* __restrict__ b1, const int* __restrict__ counts,
    const int* __restrict__ list, unsigned short* __restrict__ h) {
    const int e = blockIdx.z;
    const int tileM = blockIdx.y;
    const int count = counts[e];
    if (tileM * 16 >= count) return;          // block-uniform exit
    const int n_base = blockIdx.x * 256;

    __shared__ unsigned short ldsA[16 * kPitch];    // 16 rows x 32 K (bf16)
    __shared__ unsigned short ldsBt[256 * kPitch];  // per-n: 32 K contiguous (bf16)
    __shared__ int tok[16];
    __shared__ int rowValid[16];

    const int tid = threadIdx.x;
    if (tid < 16) {
        int p = tileM * 16 + tid;
        int q = p < count ? p : count - 1;    // clamp pad rows to a valid token
        tok[tid] = list[e * kS + q];
        rowValid[tid] = (p < count);
    }
    __syncthreads();

    const int wave = tid >> 5, lane = tid & 31;
    const int m = lane & 15, hi = lane >> 4;
    const size_t w1base = (size_t)e * kC * kH + n_base;

    v8f c0 = {}, c1 = {};
    for (int k0 = 0; k0 < kC; k0 += 32) {
        // stage A: gathered token rows, f32 -> bf16, packed-pair LDS writes
        {
            int r  = tid >> 4;                // 0..15
            int kk = (tid & 15) * 2;          // 0..30
            float2 v = *(const float2*)(x + (size_t)tok[r] * kC + k0 + kk);
            *(unsigned*)&ldsA[r * kPitch + kk] = pack2bf(v.x, v.y);
        }
        // stage B: w1[e] 32K x 256N tile -> bf16, transposed (per-n) layout
#pragma unroll
        for (int i = 0; i < 8; ++i) {
            int f  = tid + i * 256;           // float4 id, 0..2047
            int k  = f >> 6;                  // 64 float4 per K-row
            int n4 = f & 63;
            const float* src = w1 + w1base + (size_t)(k0 + k) * kH + n4 * 4;
            float4 v = *(const float4*)src;
            int nb = n4 * 4;
            ldsBt[(nb + 0) * kPitch + k] = f2bf_bits(v.x);
            ldsBt[(nb + 1) * kPitch + k] = f2bf_bits(v.y);
            ldsBt[(nb + 2) * kPitch + k] = f2bf_bits(v.z);
            ldsBt[(nb + 3) * kPitch + k] = f2bf_bits(v.w);
            if (k0 + 32 < kC)                 // global_prefetch_b8 next K-tile
                __builtin_prefetch(src + (size_t)32 * kH, 0, 1);
        }
        __syncthreads();

        // A fragment (shared by both N-subtiles): two b128 LDS loads
        v8bf a0 = *(const v8bf*)&ldsA[m * kPitch + hi * 16];       // K=hi*16+j
        v8bf a1 = *(const v8bf*)&ldsA[m * kPitch + hi * 16 + 8];   // K=hi*16+8+j
        v16bf a;
#pragma unroll
        for (int j = 0; j < 8; ++j) { a[j] = a0[j]; a[j + 8] = a1[j]; }

#pragma unroll
        for (int t = 0; t < 2; ++t) {
            int nl = wave * 32 + t * 16 + m;
            v8bf b0 = *(const v8bf*)&ldsBt[nl * kPitch + hi * 16];     // K=hi*16..+7
            v8bf b1v = *(const v8bf*)&ldsBt[nl * kPitch + hi * 16 + 8];// K=hi*16+8..
            v16bf b;
#pragma unroll
            for (int j = 0; j < 8; ++j) { b[j] = b0[j]; b[j + 8] = b1v[j]; }
            if (t == 0)
                c0 = __builtin_amdgcn_wmma_f32_16x16x32_bf16(false, a, false, b,
                                                             (short)0, c0, false, false);
            else
                c1 = __builtin_amdgcn_wmma_f32_16x16x32_bf16(false, a, false, b,
                                                             (short)0, c1, false, false);
        }
        __syncthreads();
    }

#pragma unroll
    for (int t = 0; t < 2; ++t) {
        const v8f c = t ? c1 : c0;
        const int n_out = n_base + wave * 32 + t * 16 + m;
        const float bias = b1[(size_t)e * kH + n_out];
#pragma unroll
        for (int v = 0; v < 8; ++v) {
            int row = hi * 8 + v;             // C/D layout: M = hi*8+v, N = m
            if (rowValid[row]) {
                float val = c[v] + bias;
                val = 0.5f * val * (1.0f + erff(val * 0.70710678118654752f));
                h[(size_t)tok[row] * kH + n_out] = f2bf_bits(val);
            }
        }
    }
}

// ---------------------------------------------------------------------------
// 4) fc2: out = (h @ w2[e] + b2[e]) * score. A tile (bf16 in global) is
//    fetched with global_load_async_to_lds_b128 (ASYNCcnt path); B staged
//    f32->bf16 transposed as in fc1. Block tile 16 x 256, 2 WMMA/wave/K-step.
// ---------------------------------------------------------------------------
__global__ __launch_bounds__(256) void fc2_kernel(
    const unsigned short* __restrict__ h, const float* __restrict__ w2,
    const float* __restrict__ b2, const float* __restrict__ scores,
    const int* __restrict__ counts, const int* __restrict__ list,
    float* __restrict__ out) {
    const int e = blockIdx.z;
    const int tileM = blockIdx.y;
    const int count = counts[e];
    if (tileM * 16 >= count) return;
    const int n_base = blockIdx.x * 256;

    __shared__ unsigned short ldsA[16 * kPitch];
    __shared__ unsigned short ldsBt[256 * kPitch];
    __shared__ int tok[16];
    __shared__ int rowValid[16];

    const int tid = threadIdx.x;
    if (tid < 16) {
        int p = tileM * 16 + tid;
        int q = p < count ? p : count - 1;
        tok[tid] = list[e * kS + q];
        rowValid[tid] = (p < count);
    }
    __syncthreads();

    const int wave = tid >> 5, lane = tid & 31;
    const int m = lane & 15, hi = lane >> 4;
    const size_t w2base = (size_t)e * kH * kC + n_base;
    const unsigned ldsA_base = lds_byte_off(&ldsA[0]);

    v8f c0 = {}, c1 = {};
    for (int k0 = 0; k0 < kH; k0 += 32) {
        // stage A: 16 rows x 32 bf16 via async global->LDS (wave 0 only):
        // 64 x 16B chunks, 2 instructions across 32 lanes.
        if (wave == 0) {
#pragma unroll
            for (int i = 0; i < 2; ++i) {
                int cidx = i * 32 + lane;          // 0..63
                int row = cidx >> 2, q = cidx & 3; // 4 chunks per row
                const void* g = h + (size_t)tok[row] * kH + k0 + q * 8;
                async_copy_b128(ldsA_base + row * (kPitch * 2) + q * 16, g);
            }
        }
        // stage B: w2[e] 32K x 256N tile -> bf16 transposed
#pragma unroll
        for (int i = 0; i < 8; ++i) {
            int f  = tid + i * 256;
            int k  = f >> 6;
            int n4 = f & 63;
            const float* src = w2 + w2base + (size_t)(k0 + k) * kC + n4 * 4;
            float4 v = *(const float4*)src;
            int nb = n4 * 4;
            ldsBt[(nb + 0) * kPitch + k] = f2bf_bits(v.x);
            ldsBt[(nb + 1) * kPitch + k] = f2bf_bits(v.y);
            ldsBt[(nb + 2) * kPitch + k] = f2bf_bits(v.z);
            ldsBt[(nb + 3) * kPitch + k] = f2bf_bits(v.w);
            if (k0 + 32 < kH)
                __builtin_prefetch(src + (size_t)32 * kC, 0, 1);
        }
        async_wait0();                      // drain ASYNCcnt before barrier
        __syncthreads();

        v8bf a0 = *(const v8bf*)&ldsA[m * kPitch + hi * 16];
        v8bf a1 = *(const v8bf*)&ldsA[m * kPitch + hi * 16 + 8];
        v16bf a;
#pragma unroll
        for (int j = 0; j < 8; ++j) { a[j] = a0[j]; a[j + 8] = a1[j]; }

#pragma unroll
        for (int t = 0; t < 2; ++t) {
            int nl = wave * 32 + t * 16 + m;
            v8bf b0 = *(const v8bf*)&ldsBt[nl * kPitch + hi * 16];
            v8bf b1v = *(const v8bf*)&ldsBt[nl * kPitch + hi * 16 + 8];
            v16bf b;
#pragma unroll
            for (int j = 0; j < 8; ++j) { b[j] = b0[j]; b[j + 8] = b1v[j]; }
            if (t == 0)
                c0 = __builtin_amdgcn_wmma_f32_16x16x32_bf16(false, a, false, b,
                                                             (short)0, c0, false, false);
            else
                c1 = __builtin_amdgcn_wmma_f32_16x16x32_bf16(false, a, false, b,
                                                             (short)0, c1, false, false);
        }
        __syncthreads();
    }

#pragma unroll
    for (int t = 0; t < 2; ++t) {
        const v8f c = t ? c1 : c0;
        const int n_out = n_base + wave * 32 + t * 16 + m;
        const float bias = b2[(size_t)e * kC + n_out];
#pragma unroll
        for (int v = 0; v < 8; ++v) {
            int row = hi * 8 + v;
            if (rowValid[row]) {
                int tkn = tok[row];
                out[(size_t)tkn * kC + n_out] = (c[v] + bias) * scores[tkn];
            }
        }
    }
}

// ---------------------------------------------------------------------------
// 5/6) Deterministic two-stage sum of out -> d_out[S*C]
// ---------------------------------------------------------------------------
__global__ __launch_bounds__(256) void sum1_kernel(const float* __restrict__ out,
                                                   float* __restrict__ partials) {
    __shared__ float red[256];
    int tid = threadIdx.x;
    size_t base = (size_t)blockIdx.x * 4096;
    float s = 0.f;
#pragma unroll
    for (int i = 0; i < 16; ++i) s += out[base + tid + i * 256];
    red[tid] = s;
    __syncthreads();
    for (int off = 128; off > 0; off >>= 1) {
        if (tid < off) red[tid] += red[tid + off];
        __syncthreads();
    }
    if (tid == 0) partials[blockIdx.x] = red[0];
}

__global__ __launch_bounds__(256) void sum2_kernel(const float* __restrict__ partials,
                                                   float* __restrict__ dst) {
    __shared__ float red[256];
    int tid = threadIdx.x;
    float s = partials[tid] + partials[tid + 256] + partials[tid + 512] + partials[tid + 768];
    red[tid] = s;
    __syncthreads();
    for (int off = 128; off > 0; off >>= 1) {
        if (tid < off) red[tid] += red[tid + off];
        __syncthreads();
    }
    if (tid == 0) dst[0] = red[0];
}

// ---------------------------------------------------------------------------
extern "C" void kernel_launch(void* const* d_in, const int* in_sizes, int n_in,
                              void* d_out, int out_size, void* d_ws, size_t ws_size,
                              hipStream_t stream) {
    const float* x  = (const float*)d_in[0];
    const float* wr = (const float*)d_in[1];
    const float* wg = (const float*)d_in[2];
    const float* w1 = (const float*)d_in[3];
    const float* b1 = (const float*)d_in[4];
    const float* w2 = (const float*)d_in[5];
    const float* b2 = (const float*)d_in[6];
    float* out = (float*)d_out;

    char* ws = (char*)d_ws;
    float* scores        = (float*)(ws);                       // 16 KB
    int*   eidx          = (int*)  (ws + (16u  << 10));        // 16 KB
    int*   counts        = (int*)  (ws + (32u  << 10));        // tiny
    int*   list          = (int*)  (ws + (36u  << 10));        // E*S ints = 128 KB
    float* partials      = (float*)(ws + (256u << 10));        // 4 KB
    unsigned short* hbuf = (unsigned short*)(ws + (1u << 20)); // S*H bf16 = 32 MB

    gate_kernel<<<kS / 8, 256, 0, stream>>>(x, wr, wg, scores, eidx);
    scatter_kernel<<<1, 32, 0, stream>>>(eidx, counts, list);
    fc1_kernel<<<dim3(kH / 256, kS / 16, kE), 256, 0, stream>>>(x, w1, b1, counts, list, hbuf);
    fc2_kernel<<<dim3(kC / 256, kS / 16, kE), 256, 0, stream>>>(hbuf, w2, b2, scores, counts, list, out);
    sum1_kernel<<<(kS * kC) / 4096, 256, 0, stream>>>(out, partials);
    sum2_kernel<<<1, 256, 0, stream>>>(partials, out + (size_t)kS * kC);
}
